// GATRNNAgent_46428596470411
// MI455X (gfx1250) — matrix operand order repaired and checked by reading, not compiled, not run on hardware
//
#include <hip/hip_runtime.h>
#include <hip/hip_bf16.h>

// ---------------- shapes ----------------
#define B_   4096
#define A_   32
#define OBS_ 128
#define E_   64
#define D_   64
#define H_   4
#define R_   256
#define NA_  16
#define ROWS_ (B_ * A_)          // 131072

typedef __attribute__((ext_vector_type(16))) _Float16 v16h;
typedef __attribute__((ext_vector_type(8)))  _Float16 v8h;
typedef __attribute__((ext_vector_type(8)))  float    v8f;
typedef __attribute__((ext_vector_type(4)))  float    v4f;

union H16 { v16h v; v8h h8[2]; _Float16 h[16]; };
union CFrag { v8f v; float f[8]; };

__device__ __forceinline__ v8f wmma_step(v16h a, v16h b, v8f c) {
  return __builtin_amdgcn_wmma_f32_16x16x32_f16(
      /*neg_a=*/false, a, /*neg_b=*/false, b,
      /*c_mod=*/(short)0, c, /*reuse_a=*/false, /*reuse_b=*/false);
}

// ---- fragment loaders ----
// A 16x32 f16: lane m=L&15, hf=L>>4; k run: [hf*16 .. +15] (matches swizzled B k-map)
// -> one contiguous 16-elem run; vector-loadable from row-major sources.
__device__ __forceinline__ v16h load_a_f16rm(const _Float16* __restrict__ base,
                                             int ld, int m0, int k0, int lane) {
  const int hf = lane >> 4, m = lane & 15;
  const _Float16* p = base + (size_t)(m0 + m) * ld + k0 + hf * 16;
  H16 u;
  u.h8[0] = *(const v8h*)(p);
  u.h8[1] = *(const v8h*)(p + 8);
  return u.v;
}
__device__ __forceinline__ v16h load_a_f32rm(const float* __restrict__ base,
                                             int ld, int m0, int k0, int lane) {
  const int hf = lane >> 4, m = lane & 15;
  const float* p = base + (size_t)(m0 + m) * ld + k0 + hf * 16;
  const v4f x0 = *(const v4f*)(p);
  const v4f x1 = *(const v4f*)(p + 4);
  const v4f y0 = *(const v4f*)(p + 8);
  const v4f y1 = *(const v4f*)(p + 12);
  H16 u;
#pragma unroll
  for (int e = 0; e < 4; ++e) {
    u.h[e]      = (_Float16)x0[e];
    u.h[e + 4]  = (_Float16)x1[e];
    u.h[e + 8]  = (_Float16)y0[e];
    u.h[e + 12] = (_Float16)y1[e];
  }
  return u.v;
}
// B fragments come from pre-swizzled buffers: per (ks,nt) tile, 32 lanes x 16 f16
// contiguous -> each lane reads its 32 bytes with two b128 loads.
__device__ __forceinline__ v16h load_b_sw(const _Float16* __restrict__ bsw,
                                          int ks, int Ntiles, int nt, int lane) {
  const _Float16* p = bsw + (((size_t)(ks * Ntiles + nt)) * 32 + lane) * 16;
  H16 u;
  u.h8[0] = *(const v8h*)(p);
  u.h8[1] = *(const v8h*)(p + 8);
  return u.v;
}

__device__ __forceinline__ float sigmoidf_(float x) {
  return 1.0f / (1.0f + __expf(-x));
}

// ---------------- weight swizzle kernel ----------------
// dst fragment order: [((ks*Nt + nt)*32 + lane)*16 + e]
// k = ks*32 + (lane>>4)*16 + e (B k-map); n = nt*16 + (lane&15)
// value = transpose ? src[n*ld + k] : src[k*ld + n]
__global__ void k_swizzle_b(const float* __restrict__ src, _Float16* __restrict__ dst,
                            int K, int N, int ld, int transpose) {
  const int i = blockIdx.x * blockDim.x + threadIdx.x;
  if (i >= K * N) return;
  const int e = i & 15;
  const int lane = (i >> 4) & 31;
  const int tile = i >> 9;
  const int Nt = N >> 4;
  const int ks = tile / Nt, nt = tile - ks * Nt;
  const int k = ks * 32 + ((lane >> 4) << 4) + e;
  const int n = nt * 16 + (lane & 15);
  const float v = transpose ? src[(size_t)n * ld + k] : src[(size_t)k * ld + n];
  dst[i] = (_Float16)v;
}

// ---------------- K1: GAT (one block per batch, 256 thr = 8 waves) ----------------
__global__ __launch_bounds__(256) void k_gat(
    const float* __restrict__ inputs, const int* __restrict__ mask,
    const float* __restrict__ gat_fc_b, const float* __restrict__ a_src,
    const float* __restrict__ a_dst, const float* __restrict__ b_o,
    const _Float16* __restrict__ Wg16,   // swizzled K=128,N=64
    const _Float16* __restrict__ Wh16,   // swizzled per head K=64,N=64 (4096 elems each)
    const _Float16* __restrict__ Wo16,   // swizzled K=256,N=64
    _Float16* __restrict__ att16)        // [ROWS][64]
{
  __shared__ __align__(16) _Float16 x16[A_ * E_];          // 4 KB
  __shared__ __align__(16) _Float16 h16[H_ * A_ * D_];     // 16 KB (row-major, for dots)
  __shared__ __align__(16) _Float16 hB[H_ * 4 * 32 * 16];  // 16 KB (B-fragment order)
  __shared__ float ss[H_ * A_];
  __shared__ float sd[H_ * A_];
  __shared__ __align__(16) _Float16 alpha16[H_ * A_ * A_]; // 8 KB
  __shared__ __align__(16) _Float16 aggc[A_ * (H_ * D_)];  // 16 KB

  const int b    = blockIdx.x;
  const int tid  = threadIdx.x;
  const int w    = tid >> 5;
  const int lane = tid & 31;
  const int cn   = lane & 15, chf = lane >> 4;   // C-fragment col / half
  const float* inb = inputs + (size_t)b * A_ * OBS_;

  // ---- 1) x = inputs[b] @ Wg^T + b  (32x64, K=128) ; 8 tiles, 1/wave ----
  {
    const int mt = w & 1, nt = w >> 1;
    v8f c = {};
#pragma unroll
    for (int ks = 0; ks < 4; ++ks) {
      const v16h af = load_a_f32rm(inb, OBS_, mt * 16, ks * 32, lane);
      const v16h bf = load_b_sw(Wg16, ks, 4, nt, lane);
      c = wmma_step(af, bf, c);
    }
    CFrag cf; cf.v = c;
#pragma unroll
    for (int v = 0; v < 8; ++v) {
      const int m = v + 8 * chf;
      x16[(mt * 16 + m) * E_ + nt * 16 + cn] =
          (_Float16)(cf.f[v] + gat_fc_b[nt * 16 + cn]);
    }
  }
  __syncthreads();

  // ---- 2) h[head] = x @ Wh[head]  (32x64, K=64) ; 32 tiles, 4/wave ----
#pragma unroll
  for (int t = 0; t < 4; ++t) {
    const int tile = w * 4 + t;
    const int head = tile >> 3, rem = tile & 7, mt = rem & 1, nt = rem >> 1;
    v8f c = {};
#pragma unroll
    for (int ks = 0; ks < 2; ++ks) {
      const v16h af = load_a_f16rm(x16, E_, mt * 16, ks * 32, lane);
      const v16h bf = load_b_sw(Wh16 + head * 4096, ks, 4, nt, lane);
      c = wmma_step(af, bf, c);
    }
    CFrag cf; cf.v = c;
#pragma unroll
    for (int v = 0; v < 8; ++v) {
      const int j = mt * 16 + v + 8 * chf;   // agent row
      const int d = nt * 16 + cn;            // feature col
      const _Float16 hv = (_Float16)cf.f[v];
      h16[head * (A_ * D_) + j * D_ + d] = hv;
      // also store in B-fragment order for step 5 (K=32 single k-block):
      hB[(((head * 4 + (d >> 4)) * 32) + ((j >> 4) * 16) + (d & 15)) * 16 + (j & 15)] = hv;
    }
  }
  __syncthreads();

  // ---- 3) attention scores s_src/s_dst (128 dots each, len 64) ----
  {
    const int which = tid >> 7, r = tid & 127, head = r >> 5, i = r & 31;
    const float* av = which ? a_dst : a_src;
    float s = 0.f;
    for (int d = 0; d < D_; ++d)
      s += (float)h16[head * (A_ * D_) + i * D_ + d] * av[head * D_ + d];
    if (which) sd[head * A_ + i] = s; else ss[head * A_ + i] = s;
  }
  __syncthreads();

  // ---- 4) masked leaky-relu softmax over j -> alpha ----
  if (tid < H_ * A_) {
    const int head = tid >> 5, i = tid & 31;
    const int* mrow = mask + ((size_t)b * A_ + i) * A_;
    const float si = ss[head * A_ + i];
    float ev[A_], mx = -3.0e38f;
    for (int j = 0; j < A_; ++j) {
      float v = si + sd[head * A_ + j];
      v = (v > 0.f) ? v : 0.2f * v;
      v = (mrow[j] > 0) ? v : -1.0e9f;
      ev[j] = v; mx = fmaxf(mx, v);
    }
    float sum = 0.f;
    for (int j = 0; j < A_; ++j) { float e = __expf(ev[j] - mx); ev[j] = e; sum += e; }
    const float inv = 1.0f / sum;
    for (int j = 0; j < A_; ++j)
      alpha16[head * (A_ * A_) + i * A_ + j] = (_Float16)(ev[j] * inv);
  }
  __syncthreads();

  // ---- 5) agg[head] = alpha[head] @ h[head]  (32x64, K=32) ; 32 tiles ----
#pragma unroll
  for (int t = 0; t < 4; ++t) {
    const int tile = w * 4 + t;
    const int head = tile >> 3, rem = tile & 7, mt = rem & 1, nt = rem >> 1;
    const v16h af = load_a_f16rm(alpha16 + head * (A_ * A_), A_, mt * 16, 0, lane);
    const v16h bf = load_b_sw(hB + head * 2048, 0, 4, nt, lane);
    v8f c = {};
    c = wmma_step(af, bf, c);
    CFrag cf; cf.v = c;
#pragma unroll
    for (int v = 0; v < 8; ++v) {
      const int m = v + 8 * chf;
      aggc[(mt * 16 + m) * (H_ * D_) + head * D_ + nt * 16 + cn] = (_Float16)cf.f[v];
    }
  }
  __syncthreads();

  // ---- 6) att = aggc @ Wo + b_o  (32x64, K=256) -> global f16 ----
  {
    const int mt = w & 1, nt = w >> 1;
    v8f c = {};
#pragma unroll
    for (int ks = 0; ks < 8; ++ks) {
      const v16h af = load_a_f16rm(aggc, H_ * D_, mt * 16, ks * 32, lane);
      const v16h bf = load_b_sw(Wo16, ks, 4, nt, lane);
      c = wmma_step(af, bf, c);
    }
    CFrag cf; cf.v = c;
#pragma unroll
    for (int v = 0; v < 8; ++v) {
      const int m = v + 8 * chf;
      att16[((size_t)b * A_ + mt * 16 + m) * E_ + nt * 16 + cn] =
          (_Float16)(cf.f[v] + b_o[nt * 16 + cn]);
    }
  }
}

// ---------------- K2: fc1 + GRU + LN + fc2 (one block = 16 rows) ----------------
#define K2_SMEM (16 * R_ * 2 /*xr/hn f16*/ + 16 * 3 * R_ * 4 * 2 /*gi,gh*/ + 16 * R_ * 4 /*hh*/)

__global__ __launch_bounds__(256) void k_rnn(
    const float* __restrict__ inputs, const float* __restrict__ hidden,
    const _Float16* __restrict__ att16,
    const _Float16* __restrict__ fc1T,   // swizzled K=192,N=256
    const float* __restrict__ fc1_b,
    const _Float16* __restrict__ WihT,   // swizzled K=256,N=768
    const _Float16* __restrict__ WhhT,   // swizzled K=256,N=768
    const float* __restrict__ b_ih, const float* __restrict__ b_hh,
    const float* __restrict__ ln_g, const float* __restrict__ ln_b,
    const _Float16* __restrict__ fc2T,   // swizzled K=256,N=16
    const float* __restrict__ fc2_b,
    float* __restrict__ q_out, float* __restrict__ hh_out)
{
  extern __shared__ __align__(16) char smem[];
  _Float16* xr  = (_Float16*)smem;                       // [16][256] f16 (reused as hn)
  float*    gi  = (float*)(smem + 16 * R_ * 2);          // [16][768] (reused as LN scratch)
  float*    gh  = gi + 16 * 3 * R_;                      // [16][768]
  float*    hhs = gh + 16 * 3 * R_;                      // [16][256]

  const size_t row0 = (size_t)blockIdx.x * 16;
  const int tid = threadIdx.x, w = tid >> 5, lane = tid & 31;
  const int cn = lane & 15, chf = lane >> 4;

  // ---- fc1 + ReLU: [16,192] @ [192,256] ; 16 N-tiles, 2/wave ----
#pragma unroll
  for (int t = 0; t < 2; ++t) {
    const int nt = w * 2 + t;
    v8f c = {};
#pragma unroll
    for (int ks = 0; ks < 6; ++ks) {
      const v16h af = (ks < 4)
          ? load_a_f32rm(inputs + row0 * OBS_, OBS_, 0, ks * 32, lane)
          : load_a_f16rm(att16 + row0 * E_, E_, 0, ks * 32 - OBS_, lane);
      const v16h bf = load_b_sw(fc1T, ks, 16, nt, lane);
      c = wmma_step(af, bf, c);
    }
    CFrag cf; cf.v = c;
#pragma unroll
    for (int v = 0; v < 8; ++v) {
      const int m = v + 8 * chf;
      float val = cf.f[v] + fc1_b[nt * 16 + cn];
      xr[m * R_ + nt * 16 + cn] = (_Float16)(val > 0.f ? val : 0.f);
    }
  }
  __syncthreads();

  // ---- gi = relu(fc1) @ Wih^T + b_ih ; gh = h_in @ Whh^T + b_hh  (N=768) ----
  for (int t = 0; t < 6; ++t) {
    const int nt = w * 6 + t;
    v8f c = {};
#pragma unroll
    for (int ks = 0; ks < 8; ++ks) {
      const v16h af = load_a_f16rm(xr, R_, 0, ks * 32, lane);
      const v16h bf = load_b_sw(WihT, ks, 48, nt, lane);
      c = wmma_step(af, bf, c);
    }
    CFrag cf; cf.v = c;
#pragma unroll
    for (int v = 0; v < 8; ++v) {
      const int m = v + 8 * chf;
      gi[m * (3 * R_) + nt * 16 + cn] = cf.f[v] + b_ih[nt * 16 + cn];
    }
    v8f c2 = {};
#pragma unroll
    for (int ks = 0; ks < 8; ++ks) {
      const v16h af = load_a_f32rm(hidden + row0 * R_, R_, 0, ks * 32, lane);
      const v16h bf = load_b_sw(WhhT, ks, 48, nt, lane);
      c2 = wmma_step(af, bf, c2);
    }
    cf.v = c2;
#pragma unroll
    for (int v = 0; v < 8; ++v) {
      const int m = v + 8 * chf;
      gh[m * (3 * R_) + nt * 16 + cn] = cf.f[v] + b_hh[nt * 16 + cn];
    }
  }
  __syncthreads();

  // ---- GRU elementwise (torch gate order r,z,n) ----
  for (int idx = tid; idx < 16 * R_; idx += 256) {
    const int m = idx >> 8, j = idx & (R_ - 1);
    const float gr = gi[m * 768 + j]       + gh[m * 768 + j];
    const float gz = gi[m * 768 + R_ + j]  + gh[m * 768 + R_ + j];
    const float r  = sigmoidf_(gr);
    const float zt = sigmoidf_(gz);
    const float nn = tanhf(gi[m * 768 + 2 * R_ + j] + r * gh[m * 768 + 2 * R_ + j]);
    const float hp = hidden[(row0 + m) * R_ + j];
    const float hv = (1.0f - zt) * nn + zt * hp;
    hhs[idx] = hv;
    hh_out[(row0 + m) * R_ + j] = hv;
  }
  __syncthreads();

  // ---- LayerNorm(R): 16 threads per row, sum/sumsq reduction in LDS ----
  {
    float* red = gi;  // gi dead now; reuse as scratch
    const int row = tid >> 4, t = tid & 15;
    float s1 = 0.f, s2 = 0.f;
    for (int j = t; j < R_; j += 16) {
      const float v = hhs[row * R_ + j];
      s1 += v; s2 += v * v;
    }
    red[tid] = s1; red[256 + tid] = s2;
    __syncthreads();
    if (tid < 16) {
      float a1 = 0.f, a2 = 0.f;
      for (int k = 0; k < 16; ++k) { a1 += red[tid * 16 + k]; a2 += red[256 + tid * 16 + k]; }
      const float mu = a1 * (1.0f / R_);
      const float var = a2 * (1.0f / R_) - mu * mu;
      red[512 + tid] = mu;
      red[528 + tid] = rsqrtf(var + 1e-5f);
    }
    __syncthreads();
    const float mu = red[512 + row], inv = red[528 + row];
    for (int j = t; j < R_; j += 16)
      xr[row * R_ + j] = (_Float16)((hhs[row * R_ + j] - mu) * inv * ln_g[j] + ln_b[j]);
  }
  __syncthreads();

  // ---- fc2: [16,256] @ [256,16] -> q ; single tile on wave 0 ----
  if (w == 0) {
    v8f c = {};
#pragma unroll
    for (int ks = 0; ks < 8; ++ks) {
      const v16h af = load_a_f16rm(xr, R_, 0, ks * 32, lane);
      const v16h bf = load_b_sw(fc2T, ks, 1, 0, lane);
      c = wmma_step(af, bf, c);
    }
    CFrag cf; cf.v = c;
#pragma unroll
    for (int v = 0; v < 8; ++v) {
      const int m = v + 8 * chf;
      q_out[(row0 + m) * NA_ + cn] = cf.f[v] + fc2_b[cn];
    }
  }
}

// ---------------- host launch ----------------
extern "C" void kernel_launch(void* const* d_in, const int* in_sizes, int n_in,
                              void* d_out, int out_size, void* d_ws, size_t ws_size,
                              hipStream_t stream) {
  (void)in_sizes; (void)n_in; (void)out_size; (void)ws_size;
  const float* inputs   = (const float*)d_in[0];
  const float* hidden   = (const float*)d_in[1];
  const int*   mask     = (const int*)  d_in[2];
  const float* gat_fc_w = (const float*)d_in[3];
  const float* gat_fc_b = (const float*)d_in[4];
  const float* W_h      = (const float*)d_in[5];
  const float* a_src    = (const float*)d_in[6];
  const float* a_dst    = (const float*)d_in[7];
  const float* W_o      = (const float*)d_in[8];
  const float* b_o      = (const float*)d_in[9];
  const float* fc1_w    = (const float*)d_in[10];
  const float* fc1_b    = (const float*)d_in[11];
  const float* W_ih     = (const float*)d_in[12];
  const float* W_hh     = (const float*)d_in[13];
  const float* b_ih     = (const float*)d_in[14];
  const float* b_hh     = (const float*)d_in[15];
  const float* ln_g     = (const float*)d_in[16];
  const float* ln_b     = (const float*)d_in[17];
  const float* fc2_w    = (const float*)d_in[18];
  const float* fc2_b    = (const float*)d_in[19];

  float* q_out  = (float*)d_out;                       // [ROWS][16]
  float* hh_out = q_out + (size_t)ROWS_ * NA_;         // [ROWS][256]

  // workspace layout (f16 units) — all weight buffers in swizzled fragment order
  _Float16* ws = (_Float16*)d_ws;
  _Float16* Wg16  = ws;                          // 128*64
  _Float16* Wh16  = Wg16  + 128 * 64;            // 4 * (64*64)
  _Float16* Wo16  = Wh16  + H_ * E_ * D_;        // 256*64
  _Float16* fc1T  = Wo16  + (H_ * D_) * E_;      // 192*256
  _Float16* WihT  = fc1T  + 192 * R_;            // 256*768
  _Float16* WhhT  = WihT  + R_ * 3 * R_;         // 256*768
  _Float16* fc2T  = WhhT  + R_ * 3 * R_;         // 256*16
  _Float16* att16 = fc2T  + R_ * NA_;            // ROWS*64

  const int CT = 256;
  // gat_fc_w [64][128] -> B[k][n]=w[n][k]  (transpose, ld=128)
  k_swizzle_b<<<(128 * 64 + CT - 1) / CT, CT, 0, stream>>>(gat_fc_w, Wg16, 128, 64, 128, 1);
  // W_h [head][64][64] -> direct per head (ld=64)
  for (int h = 0; h < H_; ++h)
    k_swizzle_b<<<(64 * 64 + CT - 1) / CT, CT, 0, stream>>>(
        W_h + h * E_ * D_, Wh16 + h * E_ * D_, 64, 64, 64, 0);
  // W_o [256][64] -> direct (ld=64)
  k_swizzle_b<<<(256 * 64 + CT - 1) / CT, CT, 0, stream>>>(W_o, Wo16, 256, 64, 64, 0);
  // fc1_w [256][192] -> transpose (ld=192)
  k_swizzle_b<<<(192 * 256 + CT - 1) / CT, CT, 0, stream>>>(fc1_w, fc1T, 192, 256, 192, 1);
  // W_ih / W_hh [768][256] -> transpose (ld=256)
  k_swizzle_b<<<(256 * 768 + CT - 1) / CT, CT, 0, stream>>>(W_ih, WihT, 256, 768, 256, 1);
  k_swizzle_b<<<(256 * 768 + CT - 1) / CT, CT, 0, stream>>>(W_hh, WhhT, 256, 768, 256, 1);
  // fc2_w [16][256] -> transpose (ld=256)
  k_swizzle_b<<<(256 * 16 + CT - 1) / CT, CT, 0, stream>>>(fc2_w, fc2T, 256, 16, 256, 1);

  // GAT: one block per batch element
  k_gat<<<B_, 256, 0, stream>>>(inputs, mask, gat_fc_b, a_src, a_dst, b_o,
                                Wg16, Wh16, Wo16, att16);

  // RNN: one block per 16 rows; 120 KB dynamic LDS (WGP has 320 KB)
  hipFuncSetAttribute((const void*)k_rnn,
                      hipFuncAttributeMaxDynamicSharedMemorySize, K2_SMEM);
  k_rnn<<<ROWS_ / 16, 256, K2_SMEM, stream>>>(
      inputs, hidden, att16, fc1T, fc1_b, WihT, WhhT, b_ih, b_hh,
      ln_g, ln_b, fc2T, fc2_b, q_out, hh_out);
}
